// ShiftedSWABlock_46024869544413
// MI455X (gfx1250) — matrix operand order repaired and checked by reading, not compiled, not run on hardware
//
#include <hip/hip_runtime.h>
#include <math.h>

typedef __attribute__((ext_vector_type(16))) __bf16 v16bf;
typedef __attribute__((ext_vector_type(2)))  __bf16 v2bf;
typedef __attribute__((ext_vector_type(8)))  float  v8f;

#define WMMA_BF16(a, b, c) \
  __builtin_amdgcn_wmma_f32_16x16x32_bf16(false, (a), false, (b), (short)0, (c), false, false)

// Async global->LDS DMA (CDNA5). LDS operand = low 32 bits of generic pointer
// (flat-LDS addressing uses addr[31:0] as offset into the WG allocation).
__device__ __forceinline__ void async_load_b128(const void* gaddr, void* lds) {
  unsigned l = (unsigned)(uintptr_t)lds;
  asm volatile("global_load_async_to_lds_b128 %0, %1, off"
               :: "v"(l), "v"(gaddr) : "memory");
}
#define WAIT_ASYNC(n) asm volatile("s_wait_asynccnt " #n ::: "memory")

// ---------------------------------------------------------------- constants
static constexpr int    T_TOK = 200704;        // 4*224*224 tokens
static constexpr int    NWIN  = 3136;          // total windows (4 * 784)
static constexpr size_t HW    = 224u * 224u;   // 50176

// ---------------------------------------------------------------- fragments
// A fragment (16x32 bf16, row-major, K contiguous). LDS viewed as uint pairs.
__device__ inline v16bf fragA(const unsigned* lds, int row, int pitchU, int kPairOff) {
  int lane = threadIdx.x & 31;
  const unsigned* p = lds + (size_t)(row + (lane & 15)) * pitchU + kPairOff + ((lane >> 4) << 3);
  v16bf f;
#pragma unroll
  for (int j = 0; j < 4; ++j) {
    v2bf lo = __builtin_bit_cast(v2bf, p[j]);
    v2bf hi = __builtin_bit_cast(v2bf, p[j + 4]);
    f[2*j]   = lo[0]; f[2*j+1] = lo[1];
    f[2*j+8] = hi[0]; f[2*j+9] = hi[1];
  }
  return f;
}

// B fragment for D = A * M^T where M is row-major [N,K] in LDS.
__device__ inline v16bf fragB_rows(const unsigned* lds, int nrow, int pitchU, int kPairOff) {
  int lane = threadIdx.x & 31;
  const unsigned* p = lds + (size_t)(nrow + (lane & 15)) * pitchU + kPairOff + ((lane >> 4) << 3);
  v16bf f;
#pragma unroll
  for (int j = 0; j < 8; ++j) {
    v2bf w = __builtin_bit_cast(v2bf, p[j]);
    f[2*j] = w[0]; f[2*j+1] = w[1];
  }
  return f;
}

// B fragment for D = A * M where M is row-major [K,N] (column gather).
__device__ inline v16bf fragB_cols(const __bf16* lds, int kbase, int ncol, int pitchH) {
  int lane = threadIdx.x & 31;
  int n  = ncol + (lane & 15);
  int k0 = kbase + ((lane >> 4) << 4);
  v16bf f;
#pragma unroll
  for (int j = 0; j < 8; ++j) {
    f[2*j]   = lds[(size_t)(k0 + 2*j)     * pitchH + n];
    f[2*j+1] = lds[(size_t)(k0 + 2*j + 1) * pitchH + n];
  }
  return f;
}

// ---------------------------------------------------------------- reorder in
__global__ __launch_bounds__(256) void window_kernel(const float* __restrict__ x,
                                                     float* __restrict__ xw) {
  __shared__ float tile[64][65];
  int g  = blockIdx.x;
  int b  = g / 784, m = g % 784;
  int wr = m / 28,  wc = m % 28;
  int t  = threadIdx.x;
  int n  = t & 63;
  int r  = (wr * 8 + (n >> 3) + 4) % 224;
  int c  = (wc * 8 + (n & 7)  + 4) % 224;
  size_t spat = (size_t)r * 224 + c;
  for (int cc = 0; cc < 4; ++cc) {
#pragma unroll
    for (int it = 0; it < 16; ++it) {
      int chl = (t >> 6) + it * 4;
      tile[chl][n] = x[((size_t)b * 256 + cc * 64 + chl) * HW + spat];
    }
    __syncthreads();
    int ch2 = t & 63;
#pragma unroll
    for (int it = 0; it < 16; ++it) {
      int tok = (t >> 6) + it * 4;
      xw[((size_t)g * 64 + tok) * 256 + cc * 64 + ch2] = tile[ch2][tok];
    }
    __syncthreads();
  }
}

// ---------------------------------------------------------------- reorder out
__global__ __launch_bounds__(256) void unwindow_kernel(const float* __restrict__ x3,
                                                       float* __restrict__ out) {
  __shared__ float tile[64][65];
  int g  = blockIdx.x;
  int b  = g / 784, m = g % 784;
  int wr = m / 28,  wc = m % 28;
  int t  = threadIdx.x;
  int n  = t & 63;
  int r  = (wr * 8 + (n >> 3) + 4) % 224;
  int c  = (wc * 8 + (n & 7)  + 4) % 224;
  for (int cc = 0; cc < 4; ++cc) {
    int ch2 = t & 63;
#pragma unroll
    for (int it = 0; it < 16; ++it) {
      int tok = (t >> 6) + it * 4;
      tile[ch2][tok] = x3[((size_t)g * 64 + tok) * 256 + cc * 64 + ch2];
    }
    __syncthreads();
#pragma unroll
    for (int it = 0; it < 16; ++it) {
      int chl = (t >> 6) + it * 4;
      out[((size_t)b * 256 + cc * 64 + chl) * HW + (size_t)r * 224 + c] = tile[chl][n];
    }
    __syncthreads();
  }
}

// ---------------------------------------------------------------- layernorm
__global__ __launch_bounds__(256) void ln_kernel(const float* __restrict__ x,
                                                 const float* __restrict__ w,
                                                 const float* __restrict__ b,
                                                 __bf16* __restrict__ out) {
  int wave = threadIdx.x >> 5, lane = threadIdx.x & 31;
  size_t tok = (size_t)blockIdx.x * 8 + wave;
  const float* p = x + tok * 256;
  float v[8], s = 0.f;
#pragma unroll
  for (int j = 0; j < 8; ++j) { v[j] = p[lane + 32 * j]; s += v[j]; }
#pragma unroll
  for (int off = 16; off; off >>= 1) s += __shfl_xor(s, off, 32);
  float mean = s * (1.0f / 256.0f);
  float q = 0.f;
#pragma unroll
  for (int j = 0; j < 8; ++j) { float d = v[j] - mean; q += d * d; }
#pragma unroll
  for (int off = 16; off; off >>= 1) q += __shfl_xor(q, off, 32);
  float inv = rsqrtf(q * (1.0f / 256.0f) + 1e-5f);
#pragma unroll
  for (int j = 0; j < 8; ++j) {
    int c = lane + 32 * j;
    out[tok * 256 + c] = (__bf16)((v[j] - mean) * inv * w[c] + b[c]);
  }
}

// ---------------------------------------------------------------- f32 -> bf16
__global__ void f2bf_kernel(const float* __restrict__ src, __bf16* __restrict__ dst, int n) {
  int i = blockIdx.x * 256 + threadIdx.x;
  if (i < n) dst[i] = (__bf16)src[i];
}

// ---------------------------------------------------------------- GEMM
// D[M,N] = epilogue( A[M,K] @ W[N,K]^T ), bf16 in, f32 accumulate.
// Block tile 128x256, 8 waves each 64x64 (4x4 WMMA tiles), BK=32.
// Async global->LDS DMA, double buffered; LDS pitch 20 dwords (80B, 16B-aligned).
template <bool OUT_BF16, bool HAS_BIAS, bool GELU_ACT, bool HAS_RES>
__global__ __launch_bounds__(256) void gemm_bf16(const __bf16* __restrict__ A,
                                                 const __bf16* __restrict__ W,
                                                 const float* __restrict__ bias,
                                                 const float* __restrict__ res,
                                                 __bf16* __restrict__ outb,
                                                 float* __restrict__ outf,
                                                 int M, int N, int K) {
  __shared__ __attribute__((aligned(16))) unsigned As[2][128 * 20];
  __shared__ __attribute__((aligned(16))) unsigned Bs[2][256 * 20];
  int t  = threadIdx.x;
  int bn = blockIdx.x * 256, bm = blockIdx.y * 128;
  int wave = t >> 5;
  int wm = (wave & 1) * 64, wn = (wave >> 1) * 64;

  v8f acc[4][4];
  v8f zero = {0.f, 0.f, 0.f, 0.f, 0.f, 0.f, 0.f, 0.f};
#pragma unroll
  for (int i = 0; i < 4; ++i)
#pragma unroll
    for (int j = 0; j < 4; ++j) acc[i][j] = zero;

  // per K-step each thread DMAs: A 2x16B, W 4x16B  (6 async ops / thread)
  auto issue = [&](int buf, int kk) {
#pragma unroll
    for (int i = 0; i < 2; ++i) {
      int slot = t + 256 * i;
      int row = slot >> 2, q = slot & 3;
      async_load_b128(A + (size_t)(bm + row) * K + kk + q * 8, &As[buf][row * 20 + q * 4]);
    }
#pragma unroll
    for (int i = 0; i < 4; ++i) {
      int slot = t + 256 * i;
      int row = slot >> 2, q = slot & 3;
      async_load_b128(W + (size_t)(bn + row) * K + kk + q * 8, &Bs[buf][row * 20 + q * 4]);
    }
  };

  int nsteps = K >> 5;
  issue(0, 0);
  for (int i = 0; i < nsteps; ++i) {
    int buf = i & 1;
    if (i + 1 < nsteps) {
      issue(buf ^ 1, (i + 1) << 5);
      WAIT_ASYNC(6);             // oldest batch (current buf) complete
    } else {
      WAIT_ASYNC(0);
    }
    __syncthreads();             // all waves' DMA for this buffer visible
    v16bf bf[4];
#pragma unroll
    for (int nt = 0; nt < 4; ++nt) bf[nt] = fragB_rows(Bs[buf], wn + nt * 16, 20, 0);
#pragma unroll
    for (int mt = 0; mt < 4; ++mt) {
      v16bf af = fragA(As[buf], wm + mt * 16, 20, 0);
#pragma unroll
      for (int nt = 0; nt < 4; ++nt)
        acc[mt][nt] = WMMA_BF16(af, bf[nt], acc[mt][nt]);
    }
    __syncthreads();             // done reading before next overwrite
  }

  int lane = t & 31;
  int nl = lane & 15, mo = (lane >> 4) << 3;
#pragma unroll
  for (int nt = 0; nt < 4; ++nt) {
    int gn = bn + wn + nt * 16 + nl;
    float bv = HAS_BIAS ? bias[gn] : 0.0f;
#pragma unroll
    for (int mt = 0; mt < 4; ++mt) {
#pragma unroll
      for (int r = 0; r < 8; ++r) {
        int gm = bm + wm + mt * 16 + mo + r;
        float v = acc[mt][nt][r] + bv;
        if (GELU_ACT) v = 0.5f * v * (1.0f + erff(v * 0.70710678118654752f));
        if (HAS_RES) v += res[(size_t)gm * N + gn];
        size_t o = (size_t)gm * N + gn;
        if (OUT_BF16) outb[o] = (__bf16)v;
        else          outf[o] = v;
      }
    }
  }
}

// ---------------------------------------------------------------- attention
__device__ inline int labelOf(int wr, int wc, int p) {
  int rr = wr * 8 + (p >> 3);
  int cc = wc * 8 + (p & 7);
  int a = rr < 216 ? 0 : (rr < 220 ? 1 : 2);
  int b = cc < 216 ? 0 : (cc < 220 ? 1 : 2);
  return a * 3 + b;
}

// one wave per (window g, head). qkv bf16 [T,768]; out bf16 [T,256].
__global__ __launch_bounds__(32) void attn_kernel(const __bf16* __restrict__ qkv,
                                                  const float* __restrict__ rpb,
                                                  __bf16* __restrict__ outp) {
  __shared__ __attribute__((aligned(16))) unsigned qsU[64 * 20];
  __shared__ __attribute__((aligned(16))) unsigned ksU[64 * 20];
  __shared__ __attribute__((aligned(16))) unsigned vsU[64 * 20];
  __shared__ float Ss[64 * 68];
  __shared__ unsigned PsU[64 * 34];
  __bf16* vs = (__bf16*)vsU;
  __bf16* Ps = (__bf16*)PsU;

  int task = blockIdx.x;
  int g = task >> 3, head = task & 7;
  int lane = threadIdx.x;
  int mwin = g >> 2;                 // mask index = g / B (B=4)
  int wr = mwin / 28, wc = mwin % 28;

  // DMA q,k,v (64 tokens x 32 bf16 each) into LDS
#pragma unroll
  for (int rr = 0; rr < 2; ++rr) {
    int n = lane + rr * 32;
    const __bf16* src = qkv + ((size_t)g * 64 + n) * 768 + head * 32;
#pragma unroll
    for (int j = 0; j < 4; ++j) {
      async_load_b128(src + j * 8,       &qsU[n * 20 + j * 4]);
      async_load_b128(src + 256 + j * 8, &ksU[n * 20 + j * 4]);
      async_load_b128(src + 512 + j * 8, &vsU[n * 20 + j * 4]);
    }
  }
  WAIT_ASYNC(0);
  __syncthreads();

  const float scale = 0.17677669529663687f;  // 1/sqrt(32)
  int nl = lane & 15, mo = (lane >> 4) << 3;

  // S = (q @ k^T) * scale + bias + mask
#pragma unroll
  for (int mt = 0; mt < 4; ++mt) {
    v16bf qf = fragA(qsU, mt * 16, 20, 0);
#pragma unroll
    for (int nt = 0; nt < 4; ++nt) {
      v16bf kf = fragB_rows(ksU, nt * 16, 20, 0);
      v8f s = {0.f, 0.f, 0.f, 0.f, 0.f, 0.f, 0.f, 0.f};
      s = WMMA_BF16(qf, kf, s);
      int nn = nt * 16 + nl;
      int lab_n = labelOf(wr, wc, nn);
      int yn = nn >> 3, xn = nn & 7;
#pragma unroll
      for (int r = 0; r < 8; ++r) {
        int mm = mt * 16 + mo + r;
        int ym = mm >> 3, xm = mm & 7;
        int idx = (ym - yn + 7) * 15 + (xm - xn + 7);
        float v = s[r] * scale + rpb[idx * 8 + head];
        if (labelOf(wr, wc, mm) != lab_n) v -= 100.0f;
        Ss[mm * 68 + nn] = v;
      }
    }
  }
  __syncthreads();

  // softmax over rows (2 rows per lane), write P as bf16
#pragma unroll
  for (int rr = 0; rr < 2; ++rr) {
    int m = lane + rr * 32;
    float mx = -3.4e38f;
    for (int j = 0; j < 64; ++j) mx = fmaxf(mx, Ss[m * 68 + j]);
    float sum = 0.f;
    for (int j = 0; j < 64; ++j) {
      float e = __expf(Ss[m * 68 + j] - mx);
      Ss[m * 68 + j] = e;
      sum += e;
    }
    float inv = 1.0f / sum;
    for (int j = 0; j < 64; ++j) Ps[m * 68 + j] = (__bf16)(Ss[m * 68 + j] * inv);
  }
  __syncthreads();

  // out = P @ V
#pragma unroll
  for (int mt = 0; mt < 4; ++mt) {
#pragma unroll
    for (int nt = 0; nt < 2; ++nt) {
      v8f o = {0.f, 0.f, 0.f, 0.f, 0.f, 0.f, 0.f, 0.f};
      o = WMMA_BF16(fragA(PsU, mt * 16, 34, 0),  fragB_cols(vs, 0,  nt * 16, 40), o);
      o = WMMA_BF16(fragA(PsU, mt * 16, 34, 16), fragB_cols(vs, 32, nt * 16, 40), o);
#pragma unroll
      for (int r = 0; r < 8; ++r) {
        int mm = mt * 16 + mo + r;
        outp[((size_t)g * 64 + mm) * 256 + head * 32 + nt * 16 + nl] = (__bf16)o[r];
      }
    }
  }
}

// ---------------------------------------------------------------- launch
extern "C" void kernel_launch(void* const* d_in, const int* in_sizes, int n_in,
                              void* d_out, int out_size, void* d_ws, size_t ws_size,
                              hipStream_t stream) {
  const float* x       = (const float*)d_in[0];
  const float* qkv_w   = (const float*)d_in[1];
  const float* proj_w  = (const float*)d_in[2];
  const float* proj_b  = (const float*)d_in[3];
  const float* rpb     = (const float*)d_in[4];
  const float* n1w     = (const float*)d_in[5];
  const float* n1b     = (const float*)d_in[6];
  const float* n2w     = (const float*)d_in[7];
  const float* n2b     = (const float*)d_in[8];
  const float* mlp_w1  = (const float*)d_in[9];
  const float* mlp_b1  = (const float*)d_in[10];
  const float* mlp_w2  = (const float*)d_in[11];
  const float* mlp_b2  = (const float*)d_in[12];
  float* out = (float*)d_out;

  char* ws = (char*)d_ws;
  const size_t OFF_XW  = 0;                                 // T*256 f32
  const size_t OFF_H   = OFF_XW + (size_t)T_TOK * 256 * 4;  // T*256 bf16
  const size_t OFF_QKV = OFF_H  + (size_t)T_TOK * 256 * 2;  // T*768 bf16 / reused T*256 f32
  const size_t OFF_M1  = OFF_QKV + (size_t)T_TOK * 768 * 2; // T*1024 bf16
  const size_t OFF_WB  = OFF_M1 + (size_t)T_TOK * 1024 * 2; // bf16 weights

  float*  xw   = (float*)(ws + OFF_XW);
  __bf16* hbf  = (__bf16*)(ws + OFF_H);    // LN out -> attn out -> LN2 out
  __bf16* qkvp = (__bf16*)(ws + OFF_QKV);
  float*  x2   = (float*)(ws + OFF_QKV);   // reuse after attention
  __bf16* m1p  = (__bf16*)(ws + OFF_M1);
  __bf16* wq   = (__bf16*)(ws + OFF_WB);
  __bf16* wp   = wq + 196608;
  __bf16* w1   = wp + 65536;
  __bf16* w2   = w1 + 262144;
  float*  x3   = xw;                       // reuse after proj residual consumed

  f2bf_kernel<<<(196608 + 255) / 256, 256, 0, stream>>>(qkv_w,  wq, 196608);
  f2bf_kernel<<<(65536  + 255) / 256, 256, 0, stream>>>(proj_w, wp, 65536);
  f2bf_kernel<<<(262144 + 255) / 256, 256, 0, stream>>>(mlp_w1, w1, 262144);
  f2bf_kernel<<<(262144 + 255) / 256, 256, 0, stream>>>(mlp_w2, w2, 262144);

  window_kernel<<<NWIN, 256, 0, stream>>>(x, xw);
  ln_kernel<<<T_TOK / 8, 256, 0, stream>>>(xw, n1w, n1b, hbf);
  gemm_bf16<true, false, false, false><<<dim3(3, T_TOK / 128), 256, 0, stream>>>(
      hbf, wq, nullptr, nullptr, qkvp, nullptr, T_TOK, 768, 256);
  attn_kernel<<<NWIN * 8, 32, 0, stream>>>(qkvp, rpb, hbf);
  gemm_bf16<false, true, false, true><<<dim3(1, T_TOK / 128), 256, 0, stream>>>(
      hbf, wp, proj_b, xw, nullptr, x2, T_TOK, 256, 256);
  ln_kernel<<<T_TOK / 8, 256, 0, stream>>>(x2, n2w, n2b, hbf);
  gemm_bf16<true, true, true, false><<<dim3(4, T_TOK / 128), 256, 0, stream>>>(
      hbf, w1, mlp_b1, nullptr, m1p, nullptr, T_TOK, 1024, 256);
  gemm_bf16<false, true, false, true><<<dim3(1, T_TOK / 128), 256, 0, stream>>>(
      m1p, w2, mlp_b2, x2, nullptr, x3, T_TOK, 256, 1024);
  unwindow_kernel<<<NWIN, 256, 0, stream>>>(x3, out);

  (void)in_sizes; (void)n_in; (void)out_size; (void)ws_size;
}
